// VSN_24206435680299
// MI455X (gfx1250) — matrix operand order repaired and checked by reading, not compiled, not run on hardware
//
#include <hip/hip_runtime.h>
#include <hip/hip_bf16.h>

// ---------------------------------------------------------------------------
// VSN fused forward for MI455X (gfx1250, wave32, WMMA bf16->f32)
// Round 3: native bf16 converts (v_cvt_pk_bf16_f32 path) to slim the
//          epilogue VALU; GEMM inner loop unchanged (codegen verified).
// ---------------------------------------------------------------------------

#define DEV __device__ __forceinline__

typedef __attribute__((ext_vector_type(16))) __bf16   bf16x16;
typedef __attribute__((ext_vector_type(8)))  float    f32x8;
typedef __attribute__((ext_vector_type(4)))  unsigned uint4v;

union FragU { bf16x16 v; uint4v q[2]; };

constexpr int cB = 8, cN = 384, cDIN = 40, cV = 32, cD = 256, cH = 256, cDS = 32;
constexpr int ROWS = cB * cN;        // 3072 rows (b,n)
constexpr int TR   = 128;            // rows per workgroup tile (3 blocks per batch idx)
constexpr int NRB  = ROWS / TR;      // 24 row blocks per variable
constexpr float LN_EPS = 1e-5f;
// sA (bf16) + sB (bf16) + sR (f32): 64 + 64 + 128 KB = 256 KB of 320 KB/WGP
constexpr int SMEM_BYTES = TR*cD*2 + TR*cD*2 + TR*cD*4;

// Native converts: float -> __bf16 is RNE and lowers to v_cvt_pk_bf16_f32.
DEV unsigned short f2bf(float x) {
  __bf16 b = (__bf16)x;
  return __builtin_bit_cast(unsigned short, b);
}
DEV float bf2f(unsigned short h) {
  return (float)__builtin_bit_cast(__bf16, h);
}

DEV float wred(float x) {
  #pragma unroll
  for (int o = 16; o; o >>= 1) x += __shfl_xor(x, o, 32);
  return x;
}

// A-fragment (16x32 bf16, ISA 7.12.2): lane<16 -> row=lane, K {kb..kb+7, kb+16..kb+23}
DEV FragU load_afrag(const unsigned short* ap /* row base + kt*32 */, int kb0) {
  FragU f;
  f.q[0] = *(const uint4v*)(ap + kb0);
  f.q[1] = *(const uint4v*)(ap + kb0 + 16);
  return f;
}
// B-fragment (32x16 bf16): column n=lane&15, K {kh/2..kh/2+7, kh/2+16..kh/2+23}
// (same K order as the A-fragment) -> row-major (out,in) weights load directly.
DEV FragU load_bfrag(const unsigned short* bp /* col row base + kt*32 */, int kh) {
  FragU f;
  f.q[0] = *(const uint4v*)(bp + kh / 2);
  f.q[1] = *(const uint4v*)(bp + kh / 2 + 16);
  return f;
}

// ---------------------------------------------------------------------------
// GEMM stage: OUT(128x256 bf16) = act( IN(128x256 bf16) * W^T + bias0[+bias1] )
// Wave tiling: 8 waves = 2 rowtile-blocks x 4 coltile-blocks, each 4x4 tiles.
// Per K-step: 4 A-frags (LDS) + 4 B-frags (L2) -> 16 WMMA.
// ---------------------------------------------------------------------------
DEV void gemm_stage(const unsigned short* __restrict__ sIn,
                    unsigned short* __restrict__ sOut,
                    const unsigned short* __restrict__ W,
                    const float* __restrict__ bias0,
                    const float* __restrict__ bias1,
                    bool do_elu, int tid)
{
  const int wave = tid >> 5, lane = tid & 31;
  const int rtb  = (wave >> 2) * 4;        // 0 or 4     (of 8 rowtiles)
  const int ctb  = (wave & 3) * 4;         // 0,4,8,12   (of 16 coltiles)
  const int mlo  = lane & 15;
  const int kb0  = (lane < 16) ? 0 : 8;
  const int nlo  = lane & 15;
  const int kh   = (lane < 16) ? 0 : 16;
  const int r0   = (lane < 16) ? 0 : 8;

  f32x8 acc[4][4] = {};
  #pragma unroll 1
  for (int kt = 0; kt < 8; ++kt) {
    FragU af[4], bf[4];
    #pragma unroll
    for (int r = 0; r < 4; ++r)
      af[r] = load_afrag(sIn + ((rtb + r)*16 + mlo)*cD + kt*32, kb0);
    #pragma unroll
    for (int c = 0; c < 4; ++c)
      bf[c] = load_bfrag(W + (size_t)((ctb + c)*16 + nlo)*cD + kt*32, kh);
    #pragma unroll
    for (int r = 0; r < 4; ++r)
      #pragma unroll
      for (int c = 0; c < 4; ++c)
        acc[r][c] = __builtin_amdgcn_wmma_f32_16x16x32_bf16(
            false, af[r].v, false, bf[c].v, (short)0, acc[r][c], false, false);
  }
  #pragma unroll
  for (int c = 0; c < 4; ++c) {
    const int n = (ctb + c)*16 + nlo;
    const float bsum = bias0[n] + (bias1 ? bias1[n] : 0.0f);
    #pragma unroll
    for (int r = 0; r < 4; ++r)
      #pragma unroll
      for (int i = 0; i < 8; ++i) {
        float xo = acc[r][c][i] + bsum;
        if (do_elu) xo = (xo > 0.f) ? xo : (__expf(xo) - 1.f);
        sOut[((rtb + r)*16 + r0 + i)*cD + n] = f2bf(xo);
      }
  }
}

// One 256-wide half of the gating GEMM into a 4x4 accumulator block.
DEV void gate_half(f32x8 (&acc)[4][4], const unsigned short* __restrict__ sIn,
                   const unsigned short* __restrict__ Wcol0,
                   int rtb, int ctb, int mlo, int nlo, int kb0, int kh)
{
  #pragma unroll 1
  for (int kt = 0; kt < 8; ++kt) {
    FragU af[4], bf[4];
    #pragma unroll
    for (int r = 0; r < 4; ++r)
      af[r] = load_afrag(sIn + ((rtb + r)*16 + mlo)*cD + kt*32, kb0);
    #pragma unroll
    for (int c = 0; c < 4; ++c)
      bf[c] = load_bfrag(Wcol0 + (size_t)((ctb + c)*16 + nlo)*cD + kt*32, kh);
    #pragma unroll
    for (int r = 0; r < 4; ++r)
      #pragma unroll
      for (int c = 0; c < 4; ++c)
        acc[r][c] = __builtin_amdgcn_wmma_f32_16x16x32_bf16(
            false, af[r].v, false, bf[c].v, (short)0, acc[r][c], false, false);
  }
}

// ---------------------------------------------------------------------------
// Gating stage: gl = IN * Wg^T + bgl (Wg is (512,256)); sR += a * sigmoid(g).
// Two passes to keep acc pressure at 4x4: pass G -> sigmoid parked bf16 in
// sTmp (dead ping-pong buffer, same lane rereads it), then pass A combines.
// ---------------------------------------------------------------------------
DEV void gate_stage(const unsigned short* __restrict__ sIn,
                    float* __restrict__ sR,
                    unsigned short* __restrict__ sTmp,
                    const unsigned short* __restrict__ Wg,
                    const float* __restrict__ bgl, int tid)
{
  const int wave = tid >> 5, lane = tid & 31;
  const int rtb  = (wave >> 2) * 4;
  const int ctb  = (wave & 3) * 4;
  const int mlo  = lane & 15;
  const int kb0  = (lane < 16) ? 0 : 8;
  const int nlo  = lane & 15;
  const int kh   = (lane < 16) ? 0 : 16;
  const int r0   = (lane < 16) ? 0 : 8;

  {  // ---- pass G: columns [256,512)
    f32x8 acc[4][4] = {};
    gate_half(acc, sIn, Wg + (size_t)cD * cD, rtb, ctb, mlo, nlo, kb0, kh);
    #pragma unroll
    for (int c = 0; c < 4; ++c) {
      const int na = (ctb + c)*16 + nlo;
      const float bg_ = bgl[na + cD];
      #pragma unroll
      for (int r = 0; r < 4; ++r)
        #pragma unroll
        for (int i = 0; i < 8; ++i) {
          float g = acc[r][c][i] + bg_;
          float s = 1.f / (1.f + __expf(-g));
          sTmp[((rtb + r)*16 + r0 + i)*cD + na] = f2bf(s);
        }
    }
  }
  {  // ---- pass A: columns [0,256), combine into residual
    f32x8 acc[4][4] = {};
    gate_half(acc, sIn, Wg, rtb, ctb, mlo, nlo, kb0, kh);
    #pragma unroll
    for (int c = 0; c < 4; ++c) {
      const int na = (ctb + c)*16 + nlo;
      const float ba_ = bgl[na];
      #pragma unroll
      for (int r = 0; r < 4; ++r)
        #pragma unroll
        for (int i = 0; i < 8; ++i) {
          const int rr = (rtb + r)*16 + r0 + i;
          float a = acc[r][c][i] + ba_;
          float s = bf2f(sTmp[rr*cD + na]);
          sR[rr*cD + na] += a * s;
        }
    }
  }
}

// ---------------------------------------------------------------------------
// LayerNorm over D=256; wave handles 16 rows, lane handles 8 strided columns.
// ---------------------------------------------------------------------------
DEV void ln_stage(float* __restrict__ sR, const float* __restrict__ g,
                  const float* __restrict__ be,
                  unsigned short* __restrict__ sOut,
                  unsigned short* __restrict__ gOut, int tid)
{
  const int wave = tid >> 5, lane = tid & 31;
  for (int rr = 0; rr < TR/8; ++rr) {
    const int r = wave * (TR/8) + rr;
    float vals[8], s = 0.f, s2 = 0.f;
    #pragma unroll
    for (int j = 0; j < 8; ++j) {
      float z = sR[r*cD + lane + j*32];
      vals[j] = z; s += z; s2 += z*z;
    }
    s = wred(s); s2 = wred(s2);
    const float mu  = s * (1.f / cD);
    const float inv = rsqrtf(fmaxf(s2 * (1.f / cD) - mu*mu, 0.f) + LN_EPS);
    #pragma unroll
    for (int j = 0; j < 8; ++j) {
      const int c = lane + j*32;
      float y = fmaf(g[c] * inv, vals[j] - mu, be[c]);
      sR[r*cD + c] = y;
      unsigned short h = f2bf(y);
      sOut[r*cD + c] = h;
      gOut[(size_t)r*cD + c] = h;
    }
  }
}

// Second LN fused with the scoring head: logit = grn_out . Ws + bs
DEV void ln2_logits(float* __restrict__ sR, const float* __restrict__ g,
                    const float* __restrict__ be, const float* __restrict__ Wsv,
                    float bsv, float* __restrict__ logits,
                    int row0, int v, int tid)
{
  const int wave = tid >> 5, lane = tid & 31;
  for (int rr = 0; rr < TR/8; ++rr) {
    const int r = wave * (TR/8) + rr;
    float vals[8], s = 0.f, s2 = 0.f;
    #pragma unroll
    for (int j = 0; j < 8; ++j) {
      float z = sR[r*cD + lane + j*32];
      vals[j] = z; s += z; s2 += z*z;
    }
    s = wred(s); s2 = wred(s2);
    const float mu  = s * (1.f / cD);
    const float inv = rsqrtf(fmaxf(s2 * (1.f / cD) - mu*mu, 0.f) + LN_EPS);
    float dacc = 0.f;
    #pragma unroll
    for (int j = 0; j < 8; ++j) {
      const int c = lane + j*32;
      float y = fmaf(g[c] * inv, vals[j] - mu, be[c]);
      dacc = fmaf(y, Wsv[c], dacc);
    }
    dacc = wred(dacc);
    if (lane == 0) logits[(size_t)(row0 + r)*cV + v] = dacc + bsv;
  }
}

// ---------------------------------------------------------------------------
// Main fused kernel: one block = one (variable v, 128-row tile). 8 waves.
// ---------------------------------------------------------------------------
__global__ __launch_bounds__(256)
void vsn_main(const float* __restrict__ x,   const float* __restrict__ Wp,
              const float* __restrict__ bp,  const float* __restrict__ b1s,
              const float* __restrict__ b2s, const float* __restrict__ bgs,
              const float* __restrict__ gs,  const float* __restrict__ betas,
              const float* __restrict__ b1v, const float* __restrict__ b2v,
              const float* __restrict__ bgv, const float* __restrict__ gv,
              const float* __restrict__ betav, const float* __restrict__ Ws,
              const float* __restrict__ bs,
              const unsigned short* __restrict__ W1s_b,
              const unsigned short* __restrict__ W2s_b,
              const unsigned short* __restrict__ Wgs_b,
              const unsigned short* __restrict__ W1v_b,
              const unsigned short* __restrict__ W2v_b,
              const unsigned short* __restrict__ Wgv_b,
              const float* __restrict__ ctx,
              unsigned short* __restrict__ ve_s,
              float* __restrict__ logits)
{
  extern __shared__ unsigned char smem[];
  unsigned short* sA = (unsigned short*)smem;            // 128x256 bf16
  unsigned short* sB = sA + TR*cD;                       // 128x256 bf16
  float*          sR = (float*)(sB + TR*cD);             // 128x256 f32 residual

  const int v    = blockIdx.x / NRB;   // v-major: consecutive blocks share weights in L2
  const int rb   = blockIdx.x % NRB;
  const int row0 = rb * TR;
  const int bidx = row0 / cN;          // 128 | 384 -> tile within one batch index
  const int tid  = threadIdx.x;

  // ---- stage 0: ve0 = x[:, v, None]*Wp[v] + bp[v]  (thread = column d)
  {
    const float wp  = Wp[(size_t)v*cD + tid];
    const float bpv = bp[(size_t)v*cD + tid];
    for (int m = 0; m < TR; ++m) {
      float xv  = x[(size_t)(row0 + m)*cDIN + v];
      float val = fmaf(xv, wp, bpv);
      sA[m*cD + tid] = f2bf(val);
      sR[m*cD + tid] = val;
    }
  }
  __syncthreads();

  // ---- shared GRN
  gemm_stage(sA, sB, W1s_b, b1s, nullptr, true,  tid);  __syncthreads();  // h
  gemm_stage(sB, sA, W2s_b, b2s, nullptr, false, tid);  __syncthreads();  // eta
  gate_stage(sA, sR, sB, Wgs_b, bgs, tid);              __syncthreads();  // ve0+a*sig(g)
  ln_stage(sR, gs, betas, sA,
           ve_s + ((size_t)v*ROWS + row0)*cD, tid);     __syncthreads();  // ve

  // ---- per-variable GRN
  gemm_stage(sA, sB, W1v_b + (size_t)v*cH*cD, b1v + (size_t)v*cH,
             ctx + ((size_t)bidx*cV + v)*cH, true, tid); __syncthreads(); // h2
  gemm_stage(sB, sA, W2v_b + (size_t)v*cD*cH, b2v + (size_t)v*cD,
             nullptr, false, tid);                       __syncthreads(); // eta2
  gate_stage(sA, sR, sB, Wgv_b + (size_t)v*2*cD*cD,
             bgv + (size_t)v*2*cD, tid);                 __syncthreads(); // ve+a2*sig(g2)
  ln2_logits(sR, gv + (size_t)v*cD, betav + (size_t)v*cD, Ws, bs[0],
             logits, row0, v, tid);
}

// ---------------------------------------------------------------------------
// Finalize: softmax over V (lane = variable), weighted mix of ve.
// ---------------------------------------------------------------------------
__global__ __launch_bounds__(256)
void vsn_finalize(const float* __restrict__ logits,
                  const unsigned short* __restrict__ ve_s,
                  float* __restrict__ out, float* __restrict__ wts)
{
  const int wave = threadIdx.x >> 5, lane = threadIdx.x & 31;
  const int r = blockIdx.x * 8 + wave;

  float lg = logits[(size_t)r*cV + lane];
  float mx = lg;
  #pragma unroll
  for (int o = 16; o; o >>= 1) mx = fmaxf(mx, __shfl_xor(mx, o, 32));
  float e = __expf(lg - mx);
  float ssum = e;
  #pragma unroll
  for (int o = 16; o; o >>= 1) ssum += __shfl_xor(ssum, o, 32);
  const float w = e / ssum;
  wts[(size_t)r*cV + lane] = w;

  float wv[cV];
  #pragma unroll
  for (int j = 0; j < cV; ++j) wv[j] = __shfl(w, j, 32);

  for (int d0 = 0; d0 < cD; d0 += 32) {
    const int d = d0 + lane;
    float acc = 0.f;
    #pragma unroll
    for (int vv = 0; vv < cV; ++vv)
      acc = fmaf(wv[vv], bf2f(ve_s[((size_t)vv*ROWS + r)*cD + d]), acc);
    out[(size_t)r*cD + d] = acc;
  }
}

// ---------------------------------------------------------------------------
// Prep kernels
// ---------------------------------------------------------------------------
__global__ void pack_bf16(const float* __restrict__ src,
                          unsigned short* __restrict__ dst, int n)
{
  int i = blockIdx.x * blockDim.x + threadIdx.x;
  if (i < n) dst[i] = f2bf(src[i]);
}

__global__ void ctx_kernel(const float* __restrict__ c_s,
                           const float* __restrict__ Wcv,
                           float* __restrict__ ctx)
{
  int i = blockIdx.x * blockDim.x + threadIdx.x;   // over B*V*H
  if (i >= cB * cV * cH) return;
  const int h = i % cH;
  const int v = (i / cH) % cV;
  const int b = i / (cH * cV);
  float acc = 0.f;
  #pragma unroll
  for (int s = 0; s < cDS; ++s)
    acc = fmaf(c_s[b*cDS + s], Wcv[((size_t)v*cH + h)*cDS + s], acc);
  ctx[i] = acc;   // layout [b][v][h]
}

// ---------------------------------------------------------------------------
extern "C" void kernel_launch(void* const* d_in, const int* in_sizes, int n_in,
                              void* d_out, int out_size, void* d_ws, size_t ws_size,
                              hipStream_t stream)
{
  const float* x    = (const float*)d_in[0];
  const float* c_s  = (const float*)d_in[1];
  const float* Wp   = (const float*)d_in[2];
  const float* bp   = (const float*)d_in[3];
  const float* W1s  = (const float*)d_in[4];
  const float* b1s  = (const float*)d_in[5];
  const float* W2s  = (const float*)d_in[6];
  const float* b2s  = (const float*)d_in[7];
  const float* Wgs  = (const float*)d_in[8];
  const float* bgs  = (const float*)d_in[9];
  const float* gs   = (const float*)d_in[10];
  const float* betas= (const float*)d_in[11];
  const float* W1v  = (const float*)d_in[12];
  const float* b1v  = (const float*)d_in[13];
  const float* Wcv  = (const float*)d_in[14];
  const float* W2v  = (const float*)d_in[15];
  const float* b2v  = (const float*)d_in[16];
  const float* Wgv  = (const float*)d_in[17];
  const float* bgv  = (const float*)d_in[18];
  const float* gv   = (const float*)d_in[19];
  const float* betav= (const float*)d_in[20];
  const float* Ws   = (const float*)d_in[21];
  const float* bs   = (const float*)d_in[22];

  // scratch layout
  unsigned char* p = (unsigned char*)d_ws;
  unsigned short* W1s_b = (unsigned short*)p; p += (size_t)cH*cD*2;
  unsigned short* W2s_b = (unsigned short*)p; p += (size_t)cD*cH*2;
  unsigned short* Wgs_b = (unsigned short*)p; p += (size_t)2*cD*cD*2;
  unsigned short* W1v_b = (unsigned short*)p; p += (size_t)cV*cH*cD*2;
  unsigned short* W2v_b = (unsigned short*)p; p += (size_t)cV*cD*cH*2;
  unsigned short* Wgv_b = (unsigned short*)p; p += (size_t)cV*2*cD*cD*2;
  float*          ctx   = (float*)p;          p += (size_t)cB*cV*cH*4;
  float*          logit = (float*)p;          p += (size_t)ROWS*cV*4;
  unsigned short* ve_s  = (unsigned short*)p; p += (size_t)cV*ROWS*cD*2;

  auto pk = [&](const float* s, unsigned short* d, size_t n) {
    pack_bf16<<<dim3((unsigned)((n + 255) / 256)), dim3(256), 0, stream>>>(s, d, (int)n);
  };
  pk(W1s, W1s_b, (size_t)cH*cD);
  pk(W2s, W2s_b, (size_t)cD*cH);
  pk(Wgs, Wgs_b, (size_t)2*cD*cD);
  pk(W1v, W1v_b, (size_t)cV*cH*cD);
  pk(W2v, W2v_b, (size_t)cV*cD*cH);
  pk(Wgv, Wgv_b, (size_t)cV*2*cD*cD);
  ctx_kernel<<<dim3((cB*cV*cH + 255) / 256), dim3(256), 0, stream>>>(c_s, Wcv, ctx);

  (void)hipFuncSetAttribute((const void*)vsn_main,
                            hipFuncAttributeMaxDynamicSharedMemorySize, SMEM_BYTES);

  vsn_main<<<dim3(cV * NRB), dim3(256), SMEM_BYTES, stream>>>(
      x, Wp, bp, b1s, b2s, bgs, gs, betas, b1v, b2v, bgv, gv, betav, Ws, bs,
      W1s_b, W2s_b, Wgs_b, W1v_b, W2v_b, Wgv_b, ctx, ve_s, logit);

  float* outp = (float*)d_out;
  float* wtp  = outp + (size_t)ROWS * cD;
  vsn_finalize<<<dim3(ROWS / 8), dim3(256), 0, stream>>>(logit, ve_s, outp, wtp);
}
